// LayeredGATv2_12962211299725
// MI455X (gfx1250) — compile-verified, matrix operand order checked by reading, not run on hardware
//
#include <hip/hip_runtime.h>
#include <hip/hip_bf16.h>

// ---------------- problem constants (from reference) ----------------
#define L_LAYERS 4
#define F 192          // hidden features
#define NH 3           // heads
#define HD 64          // per-head channels
#define MREAL 266      // performer random features
#define MP 288         // padded to multiple of 32 for WMMA K
#define NNODES 100000
#define TN 64          // nodes per tile
#define NT_TILES ((NNODES + TN - 1) / TN)   // 1563
#define NPAD (NT_TILES * TN)               // 100032
#define EPS_KERNEL 1e-3f
#define LN_EPS 1e-5f

#define WAVES 16
#define BLK (WAVES * 32)   // 512 threads
#define TPW 5              // ceil(72 tiles / 16 waves)

typedef __bf16 bf16_t;
typedef __attribute__((ext_vector_type(16))) __bf16 v16bf;
typedef __attribute__((ext_vector_type(8)))  __bf16 v8bf;
typedef __attribute__((ext_vector_type(8)))  float  v8f;

// ---------------- WMMA helpers (wave32, CDNA5) ----------------
__device__ inline v8f vzero8() {
  v8f z;
#pragma unroll
  for (int i = 0; i < 8; ++i) z[i] = 0.0f;
  return z;
}

// Load a 16x32 bf16 fragment. element(row r, k) = src[r*stride + k].
// A: r = M row. B: r = N column (src holds B transposed: torch [out,in] weights,
// our transposed LDS stashes). Lanes 0-15: r=lane, k in {0..7,16..23};
// lanes 16-31: r=lane-16, k in {8..15,24..31}. Two contiguous 16B loads.
__device__ inline v16bf load_frag(const bf16_t* src, int stride) {
  const int lane = threadIdx.x & 31;
  const bf16_t* p = src + (lane & 15) * stride + ((lane >> 4) << 3);
  v8bf lo = *(const v8bf*)(p);
  v8bf hi = *(const v8bf*)(p + 16);
  v16bf f;
#pragma unroll
  for (int e = 0; e < 8; ++e) { f[e] = lo[e]; f[e + 8] = hi[e]; }
  return f;
}

__device__ inline v8f wmma_bf16(v16bf a, v16bf b, v8f c) {
  return __builtin_amdgcn_wmma_f32_16x16x32_bf16(
      false, a, false, b, (short)0, c, false, false);
}

// K-loop with paired fragment loads so LDS loads of step k+1 overlap WMMA of
// step k (partial dscnt waits instead of full drains).
template <int K>
__device__ inline v8f gemm_acc(const bf16_t* A, int lda, const bf16_t* B, int ldb, v8f c) {
  int kk = 0;
#pragma unroll
  for (; kk + 64 <= K; kk += 64) {
    v16bf a0 = load_frag(A + kk, lda);
    v16bf b0 = load_frag(B + kk, ldb);
    v16bf a1 = load_frag(A + kk + 32, lda);
    v16bf b1 = load_frag(B + kk + 32, ldb);
    c = wmma_bf16(a0, b0, c);
    c = wmma_bf16(a1, b1, c);
  }
  if (kk < K) {
    v16bf a0 = load_frag(A + kk, lda);
    v16bf b0 = load_frag(B + kk, ldb);
    c = wmma_bf16(a0, b0, c);
  }
  return c;
}

// Block-cooperative global->LDS stage (16B vectors).
__device__ inline void stage_lds(bf16_t* dst, const bf16_t* src, int nelem) {
  for (int c = threadIdx.x; c < nelem / 8; c += BLK)
    *(v8bf*)(dst + c * 8) = *(const v8bf*)(src + c * 8);
}

// ---------------- conversion / utility kernels ----------------
__global__ void k_cvt_weights(const float* __restrict__ Wq, const float* __restrict__ Wk,
                              const float* __restrict__ Wv, const float* __restrict__ Wo,
                              const float* __restrict__ proj,
                              bf16_t* __restrict__ Wqb, bf16_t* __restrict__ Wkb,
                              bf16_t* __restrict__ Wvb, bf16_t* __restrict__ Wob,
                              bf16_t* __restrict__ projb) {
  int i = blockIdx.x * blockDim.x + threadIdx.x;
  const int nw = L_LAYERS * F * F;
  if (i < nw) {
    Wqb[i] = (bf16_t)Wq[i];
    Wkb[i] = (bf16_t)Wk[i];
    Wvb[i] = (bf16_t)Wv[i];
    Wob[i] = (bf16_t)Wo[i];
  }
  const int np = L_LAYERS * MP * HD;
  if (i < np) {
    int l = i / (MP * HD);
    int r = (i / HD) % MP;
    int d = i % HD;
    projb[i] = (r < MREAL) ? (bf16_t)proj[((size_t)l * MREAL + r) * HD + d] : (bf16_t)0.0f;
  }
}

__global__ void k_cvt_h(const float* __restrict__ h, bf16_t* __restrict__ hb) {
  size_t i = (size_t)blockIdx.x * blockDim.x + threadIdx.x;
  if (i < (size_t)NPAD * F) {
    size_t n = i / F;
    hb[i] = (n < NNODES) ? (bf16_t)h[i] : (bf16_t)0.0f;
  }
}

__global__ void k_zero2(float* __restrict__ p1, int n1, float* __restrict__ p2, int n2) {
  int i = blockIdx.x * blockDim.x + threadIdx.x;
  if (i < n1) p1[i] = 0.0f;
  if (i < n2) p2[i] = 0.0f;
}

__global__ void k_cvt_kv(const float* __restrict__ kv, bf16_t* __restrict__ kvT) {
  int i = blockIdx.x * blockDim.x + threadIdx.x;
  if (i < NH * MP * HD) {
    int h = i / (MP * HD);
    int m = (i / HD) % MP;
    int d = i % HD;
    kvT[((size_t)h * HD + d) * MP + m] = (bf16_t)kv[i];
  }
}

// ---------------- Phase A: kf = relu(k@projT)+eps; kv += kfT@v; ksum += sum(kf) ----------------
__global__ void __launch_bounds__(BLK, 1) k_phaseA(
    const bf16_t* __restrict__ hb, const bf16_t* __restrict__ Wkb,
    const bf16_t* __restrict__ Wvb, const bf16_t* __restrict__ projb,
    float* __restrict__ kv, float* __restrict__ ksum, int layer) {
  __shared__ __align__(16) bf16_t s_h[TN * F];      // h tile, row-major
  __shared__ __align__(16) bf16_t s_k[TN * HD];     // k head, row-major [node][d]
  __shared__ __align__(16) bf16_t s_vT[HD * TN];    // v head, transposed [d][node]
  __shared__ __align__(16) bf16_t s_kfT[MP * TN];   // kf, transposed [m][node]
  __shared__ __align__(16) bf16_t s_wk[HD * F];     // Wk head slice (LDS-resident)
  __shared__ __align__(16) bf16_t s_wv[HD * F];     // Wv head slice
  __shared__ __align__(16) bf16_t s_pj[MP * HD];    // padded proj

  const int head = blockIdx.y;
  const int wave = threadIdx.x >> 5;
  const int lane = threadIdx.x & 31;
  const int half = lane >> 4, col = lane & 15;

  float* kv_h = kv + (size_t)head * MP * HD;
  float* ks_h = ksum + (size_t)head * MP;

  // stage loop-invariant operands into LDS (keeps them off the VGPR file ->
  // no scratch spills of hoisted fragments)
  stage_lds(s_wk, Wkb + (size_t)layer * F * F + (size_t)head * HD * F, HD * F);
  stage_lds(s_wv, Wvb + (size_t)layer * F * F + (size_t)head * HD * F, HD * F);
  stage_lds(s_pj, projb + (size_t)layer * MP * HD, MP * HD);

  v8f acc[TPW];
#pragma unroll
  for (int j = 0; j < TPW; ++j) acc[j] = vzero8();
  float ksum_loc = 0.0f;

  for (int tile = blockIdx.x; tile < NT_TILES; tile += gridDim.x) {
    __syncthreads();  // previous iteration fully consumed LDS (also covers staging)
    const bf16_t* src = hb + (size_t)tile * TN * F;
    for (int c = threadIdx.x; c < (TN * F) / 8; c += BLK)
      *(v8bf*)(s_h + c * 8) = *(const v8bf*)(src + c * 8);
    if (tile + gridDim.x < NT_TILES)
      __builtin_prefetch(hb + (size_t)(tile + gridDim.x) * TN * F, 0, 1);
    __syncthreads();

    // GEMM1: k = h @ Wk_headT, v = h @ Wv_headT   (16 D-tiles, 1 per wave)
    {
      int mi = wave >> 2, ni = wave & 3;
      v8f ck = vzero8(), cv = vzero8();
      int kk = 0;
#pragma unroll
      for (; kk + 64 <= F; kk += 64) {
        v16bf a0  = load_frag(s_h + mi * 16 * F + kk, F);
        v16bf bk0 = load_frag(s_wk + ni * 16 * F + kk, F);
        v16bf bv0 = load_frag(s_wv + ni * 16 * F + kk, F);
        v16bf a1  = load_frag(s_h + mi * 16 * F + kk + 32, F);
        v16bf bk1 = load_frag(s_wk + ni * 16 * F + kk + 32, F);
        v16bf bv1 = load_frag(s_wv + ni * 16 * F + kk + 32, F);
        ck = wmma_bf16(a0, bk0, ck);
        cv = wmma_bf16(a0, bv0, cv);
        ck = wmma_bf16(a1, bk1, ck);
        cv = wmma_bf16(a1, bv1, cv);
      }
#pragma unroll
      for (int i = 0; i < 8; ++i) {
        int node = mi * 16 + i + half * 8;
        int d = ni * 16 + col;
        s_k[node * HD + d] = (bf16_t)ck[i];
        s_vT[d * TN + node] = (bf16_t)cv[i];  // store transposed
      }
    }
    __syncthreads();

    // GEMM2: kf = relu(k @ projT) + eps, stored transposed; 4x18 = 72 tiles
#pragma unroll
    for (int j = 0; j < TPW; ++j) {
      int t = j * WAVES + wave;
      if (t < 72) {
        int mi = t / 18, fi = t % 18;
        v8f c = gemm_acc<HD>(s_k + mi * 16 * HD, HD, s_pj + fi * 16 * HD, HD, vzero8());
        int mf = fi * 16 + col;
        bool mok = (mf < MREAL);
#pragma unroll
        for (int i = 0; i < 8; ++i) {
          int node = mi * 16 + i + half * 8;
          int gnode = tile * TN + node;
          float val = (mok && gnode < NNODES) ? (fmaxf(c[i], 0.0f) + EPS_KERNEL) : 0.0f;
          s_kfT[mf * TN + node] = (bf16_t)val;  // lane-contiguous 8-elem run
        }
      }
    }
    __syncthreads();

    // ksum partials: one m-row per thread
    if (threadIdx.x < MP) {
      float s = 0.0f;
      for (int n = 0; n < TN; ++n) s += (float)s_kfT[threadIdx.x * TN + n];
      ksum_loc += s;
    }

    // GEMM3: kv_acc += kfT @ v ; 18x4 = 72 tiles, persistent accumulators
#pragma unroll
    for (int j = 0; j < TPW; ++j) {
      int t = j * WAVES + wave;
      if (t < 72) {
        int mi = t >> 2, di = t & 3;
        acc[j] = gemm_acc<TN>(s_kfT + mi * 16 * TN, TN, s_vT + di * 16 * TN, TN, acc[j]);
      }
    }
  }

  // flush accumulators (f32 HW atomics)
#pragma unroll
  for (int j = 0; j < TPW; ++j) {
    int t = j * WAVES + wave;
    if (t < 72) {
      int mi = t >> 2, di = t & 3;
      int d = di * 16 + col;
#pragma unroll
      for (int i = 0; i < 8; ++i) {
        int mf = mi * 16 + i + half * 8;
        unsafeAtomicAdd(&kv_h[mf * HD + d], acc[j][i]);
      }
    }
  }
  if (threadIdx.x < MP && ksum_loc != 0.0f) unsafeAtomicAdd(&ks_h[threadIdx.x], ksum_loc);
}

// ---------------- Phase B: qf = relu(q@projT)+eps; out = (qf@kv) * (1/(qf.ksum)) ----------------
__global__ void __launch_bounds__(BLK, 1) k_phaseB(
    const bf16_t* __restrict__ hb, const bf16_t* __restrict__ Wqb,
    const bf16_t* __restrict__ projb, const bf16_t* __restrict__ kvT,
    const float* __restrict__ ksum, bf16_t* __restrict__ attn, int layer) {
  __shared__ __align__(16) bf16_t s_h[TN * F];
  __shared__ __align__(16) bf16_t s_q[TN * HD];     // row-major [node][d]
  __shared__ __align__(16) bf16_t s_qf[TN * MP];    // row-major [node][m]
  __shared__ __align__(16) bf16_t s_wq[HD * F];     // Wq head slice
  __shared__ __align__(16) bf16_t s_pj[MP * HD];    // padded proj
  __shared__ __align__(16) bf16_t s_kvT[HD * MP];   // kv^T head slice [d][m]
  __shared__ float s_red[BLK];
  __shared__ float s_dinv[TN];

  const int head = blockIdx.y;
  const int wave = threadIdx.x >> 5;
  const int lane = threadIdx.x & 31;
  const int half = lane >> 4, col = lane & 15;

  const float* ks_h = ksum + (size_t)head * MP;

  stage_lds(s_wq, Wqb + (size_t)layer * F * F + (size_t)head * HD * F, HD * F);
  stage_lds(s_pj, projb + (size_t)layer * MP * HD, MP * HD);
  stage_lds(s_kvT, kvT + (size_t)head * HD * MP, HD * MP);

  for (int tile = blockIdx.x; tile < NT_TILES; tile += gridDim.x) {
    __syncthreads();
    const bf16_t* src = hb + (size_t)tile * TN * F;
    for (int c = threadIdx.x; c < (TN * F) / 8; c += BLK)
      *(v8bf*)(s_h + c * 8) = *(const v8bf*)(src + c * 8);
    if (tile + gridDim.x < NT_TILES)
      __builtin_prefetch(hb + (size_t)(tile + gridDim.x) * TN * F, 0, 1);
    __syncthreads();

    // q = h @ Wq_headT : 16 tiles, 1 per wave
    {
      int mi = wave >> 2, ni = wave & 3;
      v8f c = gemm_acc<F>(s_h + mi * 16 * F, F, s_wq + ni * 16 * F, F, vzero8());
#pragma unroll
      for (int i = 0; i < 8; ++i) {
        int node = mi * 16 + i + half * 8;
        s_q[node * HD + ni * 16 + col] = (bf16_t)c[i];
      }
    }
    __syncthreads();

    // qf = relu(q @ projT) + eps : 72 tiles, row-major store
#pragma unroll
    for (int j = 0; j < TPW; ++j) {
      int t = j * WAVES + wave;
      if (t < 72) {
        int mi = t / 18, fi = t % 18;
        v8f c = gemm_acc<HD>(s_q + mi * 16 * HD, HD, s_pj + fi * 16 * HD, HD, vzero8());
        int mf = fi * 16 + col;
        bool mok = (mf < MREAL);
#pragma unroll
        for (int i = 0; i < 8; ++i) {
          int node = mi * 16 + i + half * 8;
          int gnode = tile * TN + node;
          float val = (mok && gnode < NNODES) ? (fmaxf(c[i], 0.0f) + EPS_KERNEL) : 0.0f;
          s_qf[node * MP + mf] = (bf16_t)val;
        }
      }
    }
    __syncthreads();

    // d = qf . ksum : 8 threads per node, LDS reduce
    {
      int node = threadIdx.x >> 3, part = threadIdx.x & 7;
      float s = 0.0f;
      int m0 = part * 34, m1 = m0 + 34 < MREAL ? m0 + 34 : MREAL;
      for (int m = m0; m < m1; ++m)
        s += (float)s_qf[node * MP + m] * ks_h[m];
      s_red[threadIdx.x] = s;
    }
    __syncthreads();
    if (threadIdx.x < TN) {
      int gnode = tile * TN + threadIdx.x;
      float s = 0.0f;
#pragma unroll
      for (int p = 0; p < 8; ++p) s += s_red[threadIdx.x * 8 + p];
      s_dinv[threadIdx.x] = (gnode < NNODES) ? 1.0f / s : 0.0f;
    }
    __syncthreads();

    // out = qf @ kv : 16 tiles, 1 per wave, K = MP
    {
      int mi = wave >> 2, ni = wave & 3;
      v8f c = gemm_acc<MP>(s_qf + mi * 16 * MP, MP, s_kvT + ni * 16 * MP, MP, vzero8());
#pragma unroll
      for (int i = 0; i < 8; ++i) {
        int node = mi * 16 + i + half * 8;
        size_t gnode = (size_t)tile * TN + node;
        float val = c[i] * s_dinv[node];
        attn[gnode * F + head * HD + ni * 16 + col] = (bf16_t)val;
      }
    }
  }
}

// ---------------- Phase C: out = attn@WoT + bo ; LayerNorm ; (ReLU) ----------------
__global__ void __launch_bounds__(BLK, 1) k_phaseC(
    const bf16_t* __restrict__ attn, const bf16_t* __restrict__ Wob,
    const float* __restrict__ bo, const float* __restrict__ gamma,
    const float* __restrict__ beta, bf16_t* __restrict__ hb_out,
    float* __restrict__ hout_f32, int layer, int apply_relu) {
  __shared__ __align__(16) bf16_t s_a[TN * F];
  __shared__ __align__(16) bf16_t s_wo[F * F];
  __shared__ float s_o[TN * F];
  __shared__ float s_rs[BLK], s_rq[BLK];
  __shared__ float s_mu[TN], s_inv[TN];

  const int tile = blockIdx.x;
  const int wave = threadIdx.x >> 5;
  const int lane = threadIdx.x & 31;
  const int half = lane >> 4, col = lane & 15;

  const float* bo_l = bo + layer * F;
  const float* g_l = gamma + layer * F;
  const float* b_l = beta + layer * F;

  stage_lds(s_wo, Wob + (size_t)layer * F * F, F * F);
  const bf16_t* src = attn + (size_t)tile * TN * F;
  for (int c = threadIdx.x; c < (TN * F) / 8; c += BLK)
    *(v8bf*)(s_a + c * 8) = *(const v8bf*)(src + c * 8);
  __syncthreads();

  // 4 x 12 = 48 output tiles, 3 per wave
#pragma unroll
  for (int j = 0; j < 3; ++j) {
    int t = j * WAVES + wave;
    int mi = t / 12, fi = t % 12;
    v8f c = gemm_acc<F>(s_a + mi * 16 * F, F, s_wo + fi * 16 * F, F, vzero8());
#pragma unroll
    for (int i = 0; i < 8; ++i) {
      int node = mi * 16 + i + half * 8;
      int f = fi * 16 + col;
      s_o[node * F + f] = c[i] + bo_l[f];
    }
  }
  __syncthreads();

  // LayerNorm stats: 8 threads per node (24 cols each), biased variance
  {
    int node = threadIdx.x >> 3, part = threadIdx.x & 7;
    const float* row = s_o + node * F;
    float sum = 0.0f, sq = 0.0f;
    for (int f = part * 24; f < part * 24 + 24; ++f) {
      float x = row[f];
      sum += x;
      sq += x * x;
    }
    s_rs[threadIdx.x] = sum;
    s_rq[threadIdx.x] = sq;
  }
  __syncthreads();
  if (threadIdx.x < TN) {
    float sum = 0.0f, sq = 0.0f;
#pragma unroll
    for (int p = 0; p < 8; ++p) {
      sum += s_rs[threadIdx.x * 8 + p];
      sq += s_rq[threadIdx.x * 8 + p];
    }
    float mu = sum * (1.0f / F);
    float var = sq * (1.0f / F) - mu * mu;
    s_mu[threadIdx.x] = mu;
    s_inv[threadIdx.x] = rsqrtf(var + LN_EPS);
  }
  __syncthreads();
  // normalize + affine (+ReLU), write bf16 next-layer h (and f32 out on last layer)
  {
    int node = threadIdx.x >> 3, part = threadIdx.x & 7;
    size_t gnode = (size_t)tile * TN + node;
    bool ok = gnode < NNODES;
    float mu = s_mu[node], inv = s_inv[node];
    const float* row = s_o + node * F;
    for (int f = part * 24; f < part * 24 + 24; ++f) {
      float y = (row[f] - mu) * inv * g_l[f] + b_l[f];
      if (apply_relu) y = fmaxf(y, 0.0f);
      hb_out[gnode * F + f] = ok ? (bf16_t)y : (bf16_t)0.0f;
      if (hout_f32 != nullptr && ok) hout_f32[gnode * F + f] = y;
    }
  }
}

// ---------------- host orchestration ----------------
extern "C" void kernel_launch(void* const* d_in, const int* in_sizes, int n_in,
                              void* d_out, int out_size, void* d_ws, size_t ws_size,
                              hipStream_t stream) {
  const float* h     = (const float*)d_in[0];
  const float* eattr = (const float*)d_in[1];
  // d_in[2] = edge_index (passthrough only, unused in compute)
  const float* Wq   = (const float*)d_in[3];
  const float* Wk   = (const float*)d_in[4];
  const float* Wv   = (const float*)d_in[5];
  const float* Wo   = (const float*)d_in[6];
  const float* bo   = (const float*)d_in[7];
  const float* gam  = (const float*)d_in[8];
  const float* bet  = (const float*)d_in[9];
  const float* proj = (const float*)d_in[10];

  uint8_t* base = (uint8_t*)d_ws;
  size_t off = 0;
  auto alloc = [&](size_t bytes) -> void* {
    void* p = base + off;
    off = (off + bytes + 255) & ~(size_t)255;
    return p;
  };
  bf16_t* hb    = (bf16_t*)alloc((size_t)NPAD * F * sizeof(bf16_t));
  bf16_t* attn  = (bf16_t*)alloc((size_t)NPAD * F * sizeof(bf16_t));
  bf16_t* Wqb   = (bf16_t*)alloc((size_t)L_LAYERS * F * F * sizeof(bf16_t));
  bf16_t* Wkb   = (bf16_t*)alloc((size_t)L_LAYERS * F * F * sizeof(bf16_t));
  bf16_t* Wvb   = (bf16_t*)alloc((size_t)L_LAYERS * F * F * sizeof(bf16_t));
  bf16_t* Wob   = (bf16_t*)alloc((size_t)L_LAYERS * F * F * sizeof(bf16_t));
  bf16_t* projb = (bf16_t*)alloc((size_t)L_LAYERS * MP * HD * sizeof(bf16_t));
  float*  kv    = (float*) alloc((size_t)NH * MP * HD * sizeof(float));
  bf16_t* kvT   = (bf16_t*)alloc((size_t)NH * HD * MP * sizeof(bf16_t));
  float*  ksum  = (float*) alloc((size_t)NH * MP * sizeof(float));

  {
    int nw = L_LAYERS * F * F, np = L_LAYERS * MP * HD;
    int ncv = nw > np ? nw : np;
    k_cvt_weights<<<(ncv + 255) / 256, 256, 0, stream>>>(Wq, Wk, Wv, Wo, proj,
                                                         Wqb, Wkb, Wvb, Wob, projb);
    size_t nh = (size_t)NPAD * F;
    k_cvt_h<<<(unsigned)((nh + 255) / 256), 256, 0, stream>>>(h, hb);
  }

  for (int l = 0; l < L_LAYERS; ++l) {
    int nz = NH * MP * HD;
    k_zero2<<<(nz + 255) / 256, 256, 0, stream>>>(kv, nz, ksum, NH * MP);
    k_phaseA<<<dim3(128, NH), BLK, 0, stream>>>(hb, Wkb, Wvb, projb, kv, ksum, l);
    k_cvt_kv<<<(NH * MP * HD + 255) / 256, 256, 0, stream>>>(kv, kvT);
    k_phaseB<<<dim3(128, NH), BLK, 0, stream>>>(hb, Wqb, projb, kvT, ksum, attn, l);
    int last = (l == L_LAYERS - 1);
    k_phaseC<<<NT_TILES, BLK, 0, stream>>>(attn, Wob, bo, gam, bet, hb,
                                           last ? (float*)d_out : (float*)nullptr,
                                           l, last ? 0 : 1);
  }

  hipMemcpyAsync((float*)d_out + (size_t)NNODES * F, eattr,
                 (size_t)in_sizes[1] * sizeof(float),
                 hipMemcpyDeviceToDevice, stream);
}
